// LATTICE_6536940224895
// MI455X (gfx1250) — matrix operand order repaired and checked by reading, not compiled, not run on hardware
//
#include <hip/hip_runtime.h>
#include <hip/hip_bf16.h>

typedef __attribute__((ext_vector_type(16))) _Float16 v16h;
typedef __attribute__((ext_vector_type(8)))  _Float16 v8h;
typedef __attribute__((ext_vector_type(4)))  _Float16 v4h;
typedef __attribute__((ext_vector_type(8)))  float    v8f;

#define N_USERS 8192
#define N_ITEMS 4096
#define DD      64
#define IMG_D   4096
#define TXT_D   384
#define TOPK    10
#define LAMBDA  0.9f
#define M_TOT   (N_USERS + N_ITEMS)
#define NNZ_E   262144
#define CHUNK   1024

__device__ __forceinline__ void modal_w(const float* mw, float& w0, float& w1) {
    float m  = fmaxf(mw[0], mw[1]);
    float e0 = __expf(mw[0] - m), e1 = __expf(mw[1] - m);
    float s  = e0 + e1;
    w0 = e0 / s; w1 = e1 / s;
}

// A-fragment (16x32 f16): lane holds row m=lane&15, K in {kh..kh+7} U {16+kh..23+kh},
// kh = 8*(lane>=16). p must point at rowBase + kb + kh (16B aligned).
__device__ __forceinline__ v16h load_a_frag(const _Float16* __restrict__ p) {
    v8h a0 = *(const v8h*)(p);
    v8h a1 = *(const v8h*)(p + 16);
    return __builtin_shufflevector(a0, a1, 0, 1, 2, 3, 4, 5, 6, 7,
                                   8, 9, 10, 11, 12, 13, 14, 15);
}

// ---------------------------------------------------------------------------
// elementwise f32 -> f16 (4 elems / thread)
// ---------------------------------------------------------------------------
__global__ void k_cvt(const float* __restrict__ src, _Float16* __restrict__ dst, int n) {
    int i = (blockIdx.x * blockDim.x + threadIdx.x) * 4;
    if (i >= n) return;
    float4 v = *(const float4*)(src + i);
    v4h h; h[0] = (_Float16)v.x; h[1] = (_Float16)v.y; h[2] = (_Float16)v.z; h[3] = (_Float16)v.w;
    *(v4h*)(dst + i) = h;
}

// f16( w0*A + w1*B ), elementwise (4 elems / thread)
__global__ void k_combine_cvt(const float* __restrict__ a, const float* __restrict__ b,
                              const float* __restrict__ mw, _Float16* __restrict__ dst, int n) {
    int i = (blockIdx.x * blockDim.x + threadIdx.x) * 4;
    if (i >= n) return;
    float w0, w1; modal_w(mw, w0, w1);
    float4 va = *(const float4*)(a + i);
    float4 vb = *(const float4*)(b + i);
    v4h h;
    h[0] = (_Float16)(w0 * va.x + w1 * vb.x);
    h[1] = (_Float16)(w0 * va.y + w1 * vb.y);
    h[2] = (_Float16)(w0 * va.z + w1 * vb.z);
    h[3] = (_Float16)(w0 * va.w + w1 * vb.w);
    *(v4h*)(dst + i) = h;
}

// src (K x 64 f32, row-major) -> dst (64 x K f16): dst[n*K + k] = src[k*64 + n]
__global__ void k_transpose_cvt(const float* __restrict__ src, _Float16* __restrict__ dst, int K) {
    int i = blockIdx.x * blockDim.x + threadIdx.x;
    if (i >= 64 * K) return;
    int n = i / K, k = i - n * K;
    dst[i] = (_Float16)src[(size_t)k * 64 + n];
}

// ---------------------------------------------------------------------------
// Y = Xh(rows x Kd) @ W(Kd x 64) + b, then row-normalize -> f16 output.
// Xh: f16 row-major; Wt: f16 64 x Kd (K-contiguous). 1 wave per 16-row tile.
// ---------------------------------------------------------------------------
__global__ void k_transform(const _Float16* __restrict__ Xh, const _Float16* __restrict__ Wt,
                            const float* __restrict__ bias, _Float16* __restrict__ Yh,
                            int Kd) {
    const int lane = threadIdx.x;          // 32 threads
    const int rb   = blockIdx.x * 16;
    const int m    = lane & 15;
    const int kh   = (lane >> 4) * 8;
    const int khB  = (lane >> 4) * 8;
    const _Float16* arow = Xh + (size_t)(rb + m) * Kd;
    v8f acc[4] = {};
    for (int kb = 0; kb < Kd; kb += 32) {
        v16h a = load_a_frag(arow + kb + kh);
#pragma unroll
        for (int t = 0; t < 4; ++t) {
            v16h bf = load_a_frag(Wt + (size_t)(t * 16 + (lane & 15)) * Kd + kb + khB);
            acc[t] = __builtin_amdgcn_wmma_f32_16x16x32_f16(false, a, false, bf,
                                                            (short)0, acc[t], false, false);
        }
    }
    __shared__ float tile[16][DD + 1];
    __shared__ float rn[16];
    const int n = lane & 15, hi = lane >> 4;
#pragma unroll
    for (int t = 0; t < 4; ++t)
#pragma unroll
        for (int v = 0; v < 8; ++v)
            tile[v + hi * 8][t * 16 + n] = acc[t][v] + bias[t * 16 + n];
    __syncthreads();
    if (lane < 16) {
        float s = 0.f;
        for (int c = 0; c < DD; ++c) { float x = tile[lane][c]; s += x * x; }
        rn[lane] = rsqrtf(s);
    }
    __syncthreads();
    for (int i = lane; i < 16 * DD; i += 32) {
        int r = i >> 6, c = i & 63;
        Yh[(size_t)(rb + r) * DD + c] = (_Float16)(tile[r][c] * rn[r]);
    }
}

// ---------------------------------------------------------------------------
// sim chunk: sim[0:CHUNK, :] = Yn[r0:r0+CHUNK] @ Yn^T  (NT, K=64)
// 8 waves / block, one 16x16 tile per wave.
// ---------------------------------------------------------------------------
__global__ void k_sim(const _Float16* __restrict__ Yh, float* __restrict__ sim, int r0) {
    const int lane = threadIdx.x & 31;
    const int wave = threadIdx.x >> 5;
    const int tile = blockIdx.x * 8 + wave;
    const int ti = tile >> 8;                  // row tile in chunk
    const int tj = tile & 255;                 // col tile
    const int m = lane & 15, kh = (lane >> 4) * 8, khB = (lane >> 4) * 8;
    const _Float16* arow = Yh + (size_t)(r0 + ti * 16 + m) * DD;
    const _Float16* brow = Yh + (size_t)(tj * 16 + (lane & 15)) * DD;
    v8f acc = {};
#pragma unroll
    for (int kb = 0; kb < DD; kb += 32) {
        v16h a  = load_a_frag(arow + kb + kh);
        v16h bf = load_a_frag(brow + kb + khB);
        acc = __builtin_amdgcn_wmma_f32_16x16x32_f16(false, a, false, bf,
                                                     (short)0, acc, false, false);
    }
    const int n = lane & 15, hi = lane >> 4;
#pragma unroll
    for (int v = 0; v < 8; ++v)
        sim[(size_t)(ti * 16 + v + hi * 8) * N_ITEMS + tj * 16 + n] = acc[v];
}

// ---------------------------------------------------------------------------
// top-10 per row of the sim chunk. 64 threads/row, register insertion lists.
// ---------------------------------------------------------------------------
__global__ void k_topk(const float* __restrict__ sim, float* __restrict__ topv,
                       int* __restrict__ topi, int r0) {
    const int r   = blockIdx.x;
    const int tid = threadIdx.x;               // 64
    const float* rowp = sim + (size_t)r * N_ITEMS;
    float tv[TOPK]; int tix[TOPK];
#pragma unroll
    for (int i = 0; i < TOPK; ++i) { tv[i] = -1e30f; tix[i] = 0; }
    for (int c = tid; c < N_ITEMS; c += 64) {
        float v = rowp[c]; int idx = c;
#pragma unroll
        for (int i = 0; i < TOPK; ++i)
            if (v > tv[i]) { float a = tv[i]; int b = tix[i]; tv[i] = v; tix[i] = idx; v = a; idx = b; }
    }
    __shared__ float sv[64 * TOPK];
    __shared__ int   si[64 * TOPK];
#pragma unroll
    for (int i = 0; i < TOPK; ++i) { sv[tid * TOPK + i] = tv[i]; si[tid * TOPK + i] = tix[i]; }
    __syncthreads();
    if (tid == 0) {
        float bv[TOPK]; int bi[TOPK];
#pragma unroll
        for (int i = 0; i < TOPK; ++i) { bv[i] = -1e30f; bi[i] = 0; }
        for (int c = 0; c < 64 * TOPK; ++c) {
            float v = sv[c]; int idx = si[c];
#pragma unroll
            for (int i = 0; i < TOPK; ++i)
                if (v > bv[i]) { float a = bv[i]; int b2 = bi[i]; bv[i] = v; bi[i] = idx; v = a; idx = b2; }
        }
        const int gr = r0 + r;
#pragma unroll
        for (int i = 0; i < TOPK; ++i) { topv[gr * TOPK + i] = bv[i]; topi[gr * TOPK + i] = bi[i]; }
    }
}

// rowsum of w0*img_adj + w1*txt_adj -> d^{-1/2}
__global__ void k_dinv(const float* __restrict__ tvi, const float* __restrict__ tvt,
                       const float* __restrict__ mw, float* __restrict__ dinv) {
    int r = blockIdx.x * blockDim.x + threadIdx.x;
    if (r >= N_ITEMS) return;
    float w0, w1; modal_w(mw, w0, w1);
    float s = 0.f;
#pragma unroll
    for (int i = 0; i < TOPK; ++i) s += w0 * tvi[r * TOPK + i] + w1 * tvt[r * TOPK + i];
    dinv[r] = (s > 0.f) ? rsqrtf(s) : 0.f;
}

// hL = learned_adj @ item_emb (20 nnz/row sparse gather)
__global__ void k_hL(const float* __restrict__ tvi, const int* __restrict__ tii,
                     const float* __restrict__ tvt, const int* __restrict__ tit,
                     const float* __restrict__ dinv, const float* __restrict__ item_emb,
                     const float* __restrict__ mw, float* __restrict__ hL) {
    const int r = blockIdx.x, d = threadIdx.x;
    float w0, w1; modal_w(mw, w0, w1);
    const float dr = dinv[r];
    float acc = 0.f;
#pragma unroll
    for (int i = 0; i < TOPK; ++i) {
        int j  = tii[r * TOPK + i];
        acc += w0 * tvi[r * TOPK + i] * dr * dinv[j]  * item_emb[(size_t)j  * DD + d];
        int j2 = tit[r * TOPK + i];
        acc += w1 * tvt[r * TOPK + i] * dr * dinv[j2] * item_emb[(size_t)j2 * DD + d];
    }
    hL[(size_t)r * DD + d] = acc;
}

// ---------------------------------------------------------------------------
// hO = Gh @ item_emb.  Gh: combined orig adj, f16 row-major 4096x4096.
// iet: item_emb transposed f16 (64 x 4096, K-contiguous).
// ---------------------------------------------------------------------------
__global__ void k_hO(const _Float16* __restrict__ Gh, const _Float16* __restrict__ iet,
                     float* __restrict__ hO) {
    const int lane = threadIdx.x;              // 32
    const int rb   = blockIdx.x * 16;
    const int m = lane & 15, kh = (lane >> 4) * 8, khB = (lane >> 4) * 8;
    const _Float16* arow = Gh + (size_t)(rb + m) * N_ITEMS;
    v8f acc[4] = {};
    for (int kb = 0; kb < N_ITEMS; kb += 32) {
        v16h a = load_a_frag(arow + kb + kh);
#pragma unroll
        for (int t = 0; t < 4; ++t) {
            v16h bf = load_a_frag(iet + (size_t)(t * 16 + (lane & 15)) * N_ITEMS + kb + khB);
            acc[t] = __builtin_amdgcn_wmma_f32_16x16x32_f16(false, a, false, bf,
                                                            (short)0, acc[t], false, false);
        }
    }
    const int n = lane & 15, hi = lane >> 4;
#pragma unroll
    for (int t = 0; t < 4; ++t)
#pragma unroll
        for (int v = 0; v < 8; ++v)
            hO[(size_t)(rb + v + hi * 8) * DD + t * 16 + n] = acc[t][v];
}

// h = 0.1*hL + 0.9*hO, then row-normalize
__global__ void k_hnorm(const float* __restrict__ hL, const float* __restrict__ hO,
                        float* __restrict__ hn) {
    const int r = blockIdx.x, d = threadIdx.x;  // 64
    float h = (1.f - LAMBDA) * hL[(size_t)r * DD + d] + LAMBDA * hO[(size_t)r * DD + d];
    __shared__ float sh[64];
    sh[d] = h * h;
    __syncthreads();
    for (int s = 32; s > 0; s >>= 1) {
        if (d < s) sh[d] += sh[d + s];
        __syncthreads();
    }
    float nrm = sqrtf(sh[0]);
    hn[(size_t)r * DD + d] = h / fmaxf(nrm, 1e-12f);
}

__global__ void k_zero(float* __restrict__ p, int n) {
    int i = blockIdx.x * blockDim.x + threadIdx.x;
    if (i < n) p[i] = 0.f;
}

// one SpMM layer via atomics; xin==nullptr means read concat(user_emb,item_emb)
__global__ void k_spmm(const int* __restrict__ rows, const int* __restrict__ cols,
                       const float* __restrict__ vals, const float* __restrict__ user_emb,
                       const float* __restrict__ item_emb, const float* __restrict__ xin,
                       float* __restrict__ out) {
    const int t = blockIdx.x * blockDim.x + threadIdx.x;
    const int e = t >> 2;
    const int q = (t & 3) * 16;
    if (e >= NNZ_E) return;
    const int c = cols[e], r = rows[e];
    const float v = vals[e];
    const float* src = xin ? (xin + (size_t)c * DD)
                           : ((c < N_USERS) ? user_emb + (size_t)c * DD
                                            : item_emb + (size_t)(c - N_USERS) * DD);
    float* dst = out + (size_t)r * DD;
#pragma unroll
    for (int d = 0; d < 16; ++d) atomicAdd(&dst[q + d], v * src[q + d]);
}

// mean of [ego0, ego1, ego2]; items += h_norm
__global__ void k_out(const float* __restrict__ user_emb, const float* __restrict__ item_emb,
                      const float* __restrict__ ego1, const float* __restrict__ ego2,
                      const float* __restrict__ hn, float* __restrict__ out) {
    int i = blockIdx.x * blockDim.x + threadIdx.x;
    if (i >= M_TOT * DD) return;
    int r = i / DD;
    float e0 = (r < N_USERS) ? user_emb[i] : item_emb[i - N_USERS * DD];
    float mean = (e0 + ego1[i] + ego2[i]) * (1.f / 3.f);
    if (r >= N_USERS) mean += hn[i - N_USERS * DD];
    out[i] = mean;
}

extern "C" void kernel_launch(void* const* d_in, const int* in_sizes, int n_in,
                              void* d_out, int out_size, void* d_ws, size_t ws_size,
                              hipStream_t stream) {
    const float* user_emb    = (const float*)d_in[0];
    const float* item_emb    = (const float*)d_in[1];
    const float* image_feats = (const float*)d_in[2];
    const float* text_feats  = (const float*)d_in[3];
    const float* image_trs_w = (const float*)d_in[4];
    const float* image_trs_b = (const float*)d_in[5];
    const float* text_trs_w  = (const float*)d_in[6];
    const float* text_trs_b  = (const float*)d_in[7];
    const float* modal_wt    = (const float*)d_in[8];
    const float* img_orig    = (const float*)d_in[9];
    const float* txt_orig    = (const float*)d_in[10];
    const float* adj_vals    = (const float*)d_in[11];
    const int*   adj_rows    = (const int*)d_in[12];
    const int*   adj_cols    = (const int*)d_in[13];
    float* out = (float*)d_out;

    char* ws = (char*)d_ws;
    size_t off = 0;
    auto alloc = [&](size_t bytes) { void* p = ws + off; off = (off + bytes + 255) & ~(size_t)255; return p; };
    _Float16*  big    = (_Float16*)alloc((size_t)N_ITEMS * IMG_D * 2);   // 32 MB, reused
    float*     simbuf = (float*)alloc((size_t)CHUNK * N_ITEMS * 4);      // 16 MB
    _Float16*  img_h  = (_Float16*)alloc((size_t)N_ITEMS * DD * 2);
    _Float16*  txt_h  = (_Float16*)alloc((size_t)N_ITEMS * DD * 2);
    _Float16*  wt_img = (_Float16*)alloc((size_t)DD * IMG_D * 2);
    _Float16*  wt_txt = (_Float16*)alloc((size_t)DD * TXT_D * 2);
    _Float16*  iet    = (_Float16*)alloc((size_t)DD * N_ITEMS * 2);
    float*     topv_i = (float*)alloc((size_t)N_ITEMS * TOPK * 4);
    int*       topi_i = (int*)alloc((size_t)N_ITEMS * TOPK * 4);
    float*     topv_t = (float*)alloc((size_t)N_ITEMS * TOPK * 4);
    int*       topi_t = (int*)alloc((size_t)N_ITEMS * TOPK * 4);
    float*     dinv   = (float*)alloc((size_t)N_ITEMS * 4);
    float*     hL     = (float*)alloc((size_t)N_ITEMS * DD * 4);
    float*     hO     = (float*)alloc((size_t)N_ITEMS * DD * 4);
    float*     hn     = (float*)alloc((size_t)N_ITEMS * DD * 4);
    float*     ego1   = (float*)alloc((size_t)M_TOT * DD * 4);
    float*     ego2   = (float*)alloc((size_t)M_TOT * DD * 4);

    // 0) operand pre-conversion (K-contiguous f16 B operands)
    k_transpose_cvt<<<(DD * IMG_D + 255) / 256, 256, 0, stream>>>(image_trs_w, wt_img, IMG_D);
    k_transpose_cvt<<<(DD * TXT_D + 255) / 256, 256, 0, stream>>>(text_trs_w, wt_txt, TXT_D);
    k_transpose_cvt<<<(DD * N_ITEMS + 255) / 256, 256, 0, stream>>>(item_emb, iet, N_ITEMS);

    // 1) modality transforms + row-normalize (f16 out); `big` holds f16 features
    const int nImg = N_ITEMS * IMG_D, nTxt = N_ITEMS * TXT_D;
    k_cvt<<<nImg / 4 / 256, 256, 0, stream>>>(image_feats, big, nImg);
    k_transform<<<N_ITEMS / 16, 32, 0, stream>>>(big, wt_img, image_trs_b, img_h, IMG_D);
    k_cvt<<<nTxt / 4 / 256, 256, 0, stream>>>(text_feats, big, nTxt);
    k_transform<<<N_ITEMS / 16, 32, 0, stream>>>(big, wt_txt, text_trs_b, txt_h, TXT_D);

    // 2) chunked sim + top-k per modality
    const int simBlocks = (CHUNK / 16) * (N_ITEMS / 16) / 8;   // 2048
    for (int c = 0; c < N_ITEMS / CHUNK; ++c) {
        k_sim <<<simBlocks, 256, 0, stream>>>(img_h, simbuf, c * CHUNK);
        k_topk<<<CHUNK, 64, 0, stream>>>(simbuf, topv_i, topi_i, c * CHUNK);
    }
    for (int c = 0; c < N_ITEMS / CHUNK; ++c) {
        k_sim <<<simBlocks, 256, 0, stream>>>(txt_h, simbuf, c * CHUNK);
        k_topk<<<CHUNK, 64, 0, stream>>>(simbuf, topv_t, topi_t, c * CHUNK);
    }

    // 3) normalized-Laplacian scaling and sparse learned-adj matvec
    k_dinv<<<(N_ITEMS + 255) / 256, 256, 0, stream>>>(topv_i, topv_t, modal_wt, dinv);
    k_hL  <<<N_ITEMS, DD, 0, stream>>>(topv_i, topi_i, topv_t, topi_t, dinv, item_emb, modal_wt, hL);

    // 4) dense original-adj GEMM (A pre-combined to f16 in `big`) and h finalize
    k_combine_cvt<<<nImg / 4 / 256, 256, 0, stream>>>(img_orig, txt_orig, modal_wt, big, nImg);
    k_hO   <<<N_ITEMS / 16, 32, 0, stream>>>(big, iet, hO);
    k_hnorm<<<N_ITEMS, DD, 0, stream>>>(hL, hO, hn);

    // 5) two UI SpMM layers
    const int egoN = M_TOT * DD;
    k_zero<<<(egoN + 255) / 256, 256, 0, stream>>>(ego1, egoN);
    k_spmm<<<(NNZ_E * 4) / 256, 256, 0, stream>>>(adj_rows, adj_cols, adj_vals,
                                                  user_emb, item_emb, nullptr, ego1);
    k_zero<<<(egoN + 255) / 256, 256, 0, stream>>>(ego2, egoN);
    k_spmm<<<(NNZ_E * 4) / 256, 256, 0, stream>>>(adj_rows, adj_cols, adj_vals,
                                                  user_emb, item_emb, ego1, ego2);

    // 6) epilogue -> d_out = [u_g ; i_g + h_norm]
    k_out<<<(egoN + 255) / 256, 256, 0, stream>>>(user_emb, item_emb, ego1, ego2, hn, out);
    (void)in_sizes; (void)n_in; (void)out_size; (void)ws_size;
}
